// QuantizedLinearFSDP_50165218017637
// MI455X (gfx1250) — matrix operand order repaired and checked by reading, not compiled, not run on hardware
//
#include <hip/hip_runtime.h>

// ---------------------------------------------------------------------------
// Quantized (codebook) linear: out[M,N] = x[M,K] * W[N,K]^T + bias
//   W[o, g*128+s] = codebooks[o, g, codes[o,g,s]]
// M=2048, N=4096, K=4096, 32 groups of 128, 16-entry codebooks.
//
// LDS-tiled bf16 WMMA GEMM with fp32-class accuracy via hi/lo bf16 split
// (3 WMMAs: ah*bh + ah*bl + al*bh).
//  - Pre-pass kernel splits x -> xh/xl bf16 planes ONCE (workspace).
//  - A-tile fill uses gfx1250 GLOBAL_LOAD_ASYNC_TO_LDS_B128 (ASYNCcnt),
//    bypassing VGPRs entirely; completion via s_wait_asynccnt.
//  - Codebook entries split ONCE PER GROUP into LDS, packed hi|lo<<16;
//    dequant is 1 ds_load_b32 + unpack per weight element.
// ---------------------------------------------------------------------------

typedef __attribute__((ext_vector_type(16))) __bf16 v16bf;
typedef __attribute__((ext_vector_type(8)))  float  v8f;

#define TM     128
#define TN     128
#define TK     32
#define GROUPS 32
#define GSIZE  128
#define CBS    16
#define KDIM   (GROUPS * GSIZE)  // 4096
#define NDIM   4096
#define LDT    40    // padded LDS row length in ushort (80B, 16B aligned)

__device__ __forceinline__ void split_bf(float f, unsigned short& h, unsigned short& l) {
  unsigned int u = __float_as_uint(f);
  unsigned int r = u + (0x7FFFu + ((u >> 16) & 1u));   // RNE to bf16
  h = (unsigned short)(r >> 16);
  float hf = __uint_as_float((unsigned int)h << 16);
  float lf = f - hf;
  unsigned int u2 = __float_as_uint(lf);
  l = (unsigned short)((u2 + (0x7FFFu + ((u2 >> 16) & 1u))) >> 16);
}

// low 32 bits of a generic LDS pointer == logical LDS byte address (ISA 10.2)
__device__ __forceinline__ unsigned lds_addr_of(const void* p) {
  return (unsigned)(size_t)p;
}

union FragU { uint4 q[2]; v16bf v; };

// ---------------- pre-pass: split x into hi/lo bf16 planes ------------------
__global__ __launch_bounds__(256)
void split_x_kernel(const float* __restrict__ x,
                    unsigned short* __restrict__ xh,
                    unsigned short* __restrict__ xl, long n) {
  long i = ((long)blockIdx.x * blockDim.x + threadIdx.x) * 8;
  if (i >= n) return;
  float4 a = *(const float4*)(x + i);
  float4 b = *(const float4*)(x + i + 4);
  unsigned short th[8], tl[8];
  split_bf(a.x, th[0], tl[0]); split_bf(a.y, th[1], tl[1]);
  split_bf(a.z, th[2], tl[2]); split_bf(a.w, th[3], tl[3]);
  split_bf(b.x, th[4], tl[4]); split_bf(b.y, th[5], tl[5]);
  split_bf(b.z, th[6], tl[6]); split_bf(b.w, th[7], tl[7]);
  *(uint4*)(xh + i) = *(const uint4*)(&th[0]);
  *(uint4*)(xl + i) = *(const uint4*)(&tl[0]);
}

// ---------------- main GEMM -------------------------------------------------
template <bool PRESPLIT>
__global__ __launch_bounds__(256)
void aqlm_wmma_gemm(const unsigned short* __restrict__ xh,
                    const unsigned short* __restrict__ xl,
                    const float* __restrict__ x,
                    const float* __restrict__ codebooks,
                    const int*   __restrict__ codes,
                    const float* __restrict__ bias,
                    float* __restrict__ out, int M) {
  __shared__ __align__(16) unsigned short lAh[TM * LDT];
  __shared__ __align__(16) unsigned short lAl[TM * LDT];
  __shared__ __align__(16) unsigned short lBh[TN * LDT];
  __shared__ __align__(16) unsigned short lBl[TN * LDT];
  __shared__ __align__(16) unsigned int   lCBp[TN * CBS];  // hi | lo<<16

  const int t    = threadIdx.x;
  const int lane = t & 31;
  const int wave = t >> 5;
  const int n0   = blockIdx.x * TN;
  const int m0   = blockIdx.y * TM;
  const int wm0  = (wave >> 2) * 64;
  const int wn0  = (wave & 3) * 32;

  v8f acc[4][2];
#pragma unroll
  for (int i = 0; i < 4; ++i)
#pragma unroll
    for (int j = 0; j < 2; ++j)
      acc[i][j] = (v8f){0.f, 0.f, 0.f, 0.f, 0.f, 0.f, 0.f, 0.f};

  const int fr  = t >> 1;          // tile row 0..127 (both A and B fills)
  const int fc0 = (t & 1) * 16;    // col offset 0 or 16

  // per-thread LDS byte addresses for async A-tile fill
  const unsigned ldsAh = lds_addr_of(&lAh[fr * LDT + fc0]);
  const unsigned ldsAl = lds_addr_of(&lAl[fr * LDT + fc0]);

  for (int g = 0; g < GROUPS; ++g) {
    // ---- stage + split codebook slice for this group (amortized over 4 ks) --
    {
      const int o  = t >> 1;
      const int j0 = (t & 1) * 8;
      const float* src = codebooks + ((long)(n0 + o) * GROUPS + g) * CBS + j0;
      float4 c0 = *(const float4*)(src);
      float4 c1 = *(const float4*)(src + 4);
      float v[8] = {c0.x, c0.y, c0.z, c0.w, c1.x, c1.y, c1.z, c1.w};
      unsigned int* dst = &lCBp[o * CBS + j0];
#pragma unroll
      for (int j = 0; j < 8; ++j) {
        unsigned short h, l;
        split_bf(v[j], h, l);
        dst[j] = (unsigned int)h | ((unsigned int)l << 16);
      }
    }
    __syncthreads();

#pragma unroll 1
    for (int ks = 0; ks < GSIZE / TK; ++ks) {
      const int k0 = g * GSIZE + ks * TK;
      const int s0 = ks * TK;

      // ---- A tile fill -----------------------------------------------------
      if (PRESPLIT) {
        // direct global -> LDS async copy (ASYNCcnt), 32B per plane per thread
        const unsigned goff =
            (unsigned)(((long)(m0 + fr) * KDIM + k0 + fc0) * sizeof(unsigned short));
        asm volatile(
            "global_load_async_to_lds_b128 %0, %2, %3 offset:0\n\t"
            "global_load_async_to_lds_b128 %0, %2, %3 offset:16\n\t"
            "global_load_async_to_lds_b128 %1, %2, %4 offset:0\n\t"
            "global_load_async_to_lds_b128 %1, %2, %4 offset:16"
            :
            : "v"(ldsAh), "v"(ldsAl), "v"(goff), "s"(xh), "s"(xl)
            : "memory");
        __builtin_prefetch(xh + (long)(m0 + fr) * KDIM + k0 + fc0 + TK, 0, 1);
      } else {
        const float* src = x + (long)(m0 + fr) * KDIM + k0 + fc0;
        unsigned short th[16], tl[16];
#pragma unroll
        for (int q = 0; q < 4; ++q) {
          float4 v = *(const float4*)(src + q * 4);
          split_bf(v.x, th[q*4+0], tl[q*4+0]);
          split_bf(v.y, th[q*4+1], tl[q*4+1]);
          split_bf(v.z, th[q*4+2], tl[q*4+2]);
          split_bf(v.w, th[q*4+3], tl[q*4+3]);
        }
        unsigned short* dh = &lAh[fr * LDT + fc0];
        unsigned short* dl = &lAl[fr * LDT + fc0];
        *(uint4*)(dh)     = *(const uint4*)(&th[0]);
        *(uint4*)(dh + 8) = *(const uint4*)(&th[8]);
        *(uint4*)(dl)     = *(const uint4*)(&tl[0]);
        *(uint4*)(dl + 8) = *(const uint4*)(&tl[8]);
        __builtin_prefetch(src + TK, 0, 1);
      }
      // ---- B tile fill: gather pre-split packed codebook -------------------
      {
        const int* csrc = codes + ((long)(n0 + fr) * GROUPS + g) * GSIZE + s0 + fc0;
        const unsigned int* cbp = &lCBp[fr * CBS];
        unsigned short th[16], tl[16];
#pragma unroll
        for (int q = 0; q < 4; ++q) {
          int4 c4 = *(const int4*)(csrc + q * 4);
          unsigned int p0 = cbp[c4.x & 15];
          unsigned int p1 = cbp[c4.y & 15];
          unsigned int p2 = cbp[c4.z & 15];
          unsigned int p3 = cbp[c4.w & 15];
          th[q*4+0] = (unsigned short)p0;  tl[q*4+0] = (unsigned short)(p0 >> 16);
          th[q*4+1] = (unsigned short)p1;  tl[q*4+1] = (unsigned short)(p1 >> 16);
          th[q*4+2] = (unsigned short)p2;  tl[q*4+2] = (unsigned short)(p2 >> 16);
          th[q*4+3] = (unsigned short)p3;  tl[q*4+3] = (unsigned short)(p3 >> 16);
        }
        unsigned short* dh = &lBh[fr * LDT + fc0];
        unsigned short* dl = &lBl[fr * LDT + fc0];
        *(uint4*)(dh)     = *(const uint4*)(&th[0]);
        *(uint4*)(dh + 8) = *(const uint4*)(&th[8]);
        *(uint4*)(dl)     = *(const uint4*)(&tl[0]);
        *(uint4*)(dl + 8) = *(const uint4*)(&tl[8]);
        __builtin_prefetch(csrc + TK, 0, 1);
      }
      if (PRESPLIT) {
        // async LDS writes must land before the consumer barrier
        asm volatile("s_wait_asynccnt 0x0" ::: "memory");
      }
      __syncthreads();

      // ---- fragments + WMMA ------------------------------------------------
      const int hr = lane & 15;
      const int hi = lane >> 4;
      FragU aH[4], aL[4];
#pragma unroll
      for (int fm = 0; fm < 4; ++fm) {
        const int row = (wm0 + fm * 16 + hr) * LDT;
        aH[fm].q[0] = *(const uint4*)(&lAh[row + hi * 8]);
        aH[fm].q[1] = *(const uint4*)(&lAh[row + 16 + hi * 8]);
        aL[fm].q[0] = *(const uint4*)(&lAl[row + hi * 8]);
        aL[fm].q[1] = *(const uint4*)(&lAl[row + 16 + hi * 8]);
      }
      FragU bH[2], bL[2];
#pragma unroll
      for (int fn = 0; fn < 2; ++fn) {
        const int row = (wn0 + fn * 16 + hr) * LDT + hi * 8;
        bH[fn].q[0] = *(const uint4*)(&lBh[row]);
        bH[fn].q[1] = *(const uint4*)(&lBh[row + 16]);
        bL[fn].q[0] = *(const uint4*)(&lBl[row]);
        bL[fn].q[1] = *(const uint4*)(&lBl[row + 16]);
      }
#pragma unroll
      for (int fm = 0; fm < 4; ++fm)
#pragma unroll
        for (int fn = 0; fn < 2; ++fn) {
          acc[fm][fn] = __builtin_amdgcn_wmma_f32_16x16x32_bf16(
              false, aH[fm].v, false, bH[fn].v, (short)0, acc[fm][fn], false, false);
          acc[fm][fn] = __builtin_amdgcn_wmma_f32_16x16x32_bf16(
              false, aH[fm].v, false, bL[fn].v, (short)0, acc[fm][fn], false, false);
          acc[fm][fn] = __builtin_amdgcn_wmma_f32_16x16x32_bf16(
              false, aL[fm].v, false, bH[fn].v, (short)0, acc[fm][fn], false, false);
        }
      __syncthreads();
    }
  }

  // ---- epilogue: D layout lane<16 -> M=v, N=lane ; lane>=16 -> M=8+v -------
  const int hr = lane & 15;
  const int hi = lane >> 4;
#pragma unroll
  for (int fn = 0; fn < 2; ++fn) {
    const int n  = n0 + wn0 + fn * 16 + hr;
    const float bv = bias[n];
#pragma unroll
    for (int fm = 0; fm < 4; ++fm) {
      const int mbase = m0 + wm0 + fm * 16 + hi * 8;
#pragma unroll
      for (int v = 0; v < 8; ++v)
        out[(long)(mbase + v) * NDIM + n] = acc[fm][fn][v] + bv;
    }
  }
}

extern "C" void kernel_launch(void* const* d_in, const int* in_sizes, int n_in,
                              void* d_out, int out_size, void* d_ws, size_t ws_size,
                              hipStream_t stream) {
  const float* x         = (const float*)d_in[0];
  const float* codebooks = (const float*)d_in[1];
  const int*   codes     = (const int*)d_in[2];
  const float* bias      = (const float*)d_in[3];
  float*       out       = (float*)d_out;

  const int  M      = in_sizes[0] / KDIM;          // 2048
  const long xElems = (long)M * KDIM;              // 8.4M
  const size_t need = (size_t)xElems * 2 * sizeof(unsigned short);

  dim3 grid(NDIM / TN, M / TM);                    // (32, 16)

  if (ws_size >= need) {
    unsigned short* xh = (unsigned short*)d_ws;
    unsigned short* xl = xh + xElems;
    long nvec = xElems / 8;                        // 8 elements per thread
    split_x_kernel<<<(nvec + 255) / 256, 256, 0, stream>>>(x, xh, xl, xElems);
    aqlm_wmma_gemm<true><<<grid, 256, 0, stream>>>(xh, xl, x, codebooks, codes,
                                                   bias, out, M);
  } else {
    aqlm_wmma_gemm<false><<<grid, 256, 0, stream>>>(nullptr, nullptr, x,
                                                    codebooks, codes, bias, out, M);
  }
}
